// LightGGNNEnhancer_9165460209924
// MI455X (gfx1250) — compile-verified, matrix operand order checked by reading, not run on hardware
//
#include <hip/hip_runtime.h>
#include <math.h>

typedef __attribute__((ext_vector_type(16))) _Float16 v16h;
typedef __attribute__((ext_vector_type(8)))  float    v8f;

#define NEG_SLOPE 0.2f

// ---------------------------------------------------------------------------
// float atomic max via orderable-int trick (init memory to -inf)
// ---------------------------------------------------------------------------
static __device__ __forceinline__ void atomicMaxF(float* addr, float val) {
  if (val >= 0.0f) atomicMax((int*)addr, __float_as_int(val));
  else             atomicMin((unsigned int*)addr, __float_as_uint(val));
}

__global__ void fill_kernel(float* __restrict__ p, int n, float v) {
  int i = blockIdx.x * blockDim.x + threadIdx.x;
  if (i < n) p[i] = v;
}

// ---------------------------------------------------------------------------
// Pre-swizzle W[K x NOUT] (f32 row-major) into WMMA B-fragment order (f16):
// out[((ks32*NT + nt)*32 + lane)*16 + elem]
//   lane: koff = 8*(lane>>4), ncol = nt*16 + (lane&15)
//   elem t   (0..7):  K = ks32*32 + koff + t
//   elem 8+t (0..7):  K = ks32*32 + 16 + koff + t
// so the GEMM can load each fragment as one contiguous v16h (2x b128).
// ---------------------------------------------------------------------------
__global__ void swizzle_w_kernel(const float* __restrict__ W, _Float16* __restrict__ out,
                                 int K, int NOUT) {
  int i = blockIdx.x * blockDim.x + threadIdx.x;
  int NT = NOUT >> 4;
  int total = (K >> 5) * NT * 32 * 16;
  if (i >= total) return;
  int elem = i & 15;
  int lane = (i >> 4) & 31;
  int rest = i >> 9;
  int nt   = rest % NT;
  int ks32 = rest / NT;
  int koff = (lane >> 4) * 8;
  int ncol = nt * 16 + (lane & 15);
  int t    = elem & 7;
  int khi  = (elem >> 3) * 16;
  int k    = ks32 * 32 + khi + koff + t;
  out[i] = (_Float16)W[k * NOUT + ncol];
}

// ---------------------------------------------------------------------------
// Tall-skinny GEMM  Y[N x NOUT] = X[N x K] @ W[K x NOUT]  (f16 WMMA, f32 acc)
// One wave per 16-row tile. B comes pre-swizzled (see above); A is loaded as
// four float4s per k-step and packed to f16.
// C layout: VGPR r -> row r + 8*(lane>>4); lane -> column (lane&15).
// ---------------------------------------------------------------------------
template<int K, int NOUT>
__global__ void gemm_wmma_kernel(const float* __restrict__ X,
                                 const _Float16* __restrict__ Wswz,
                                 float* __restrict__ Y, int Nn) {
  constexpr int NT = NOUT / 16;
  const int lane = threadIdx.x;          // 0..31
  const int row0 = blockIdx.x * 16;
  const int mrel = lane & 15;
  const int grp  = lane >> 4;            // 0 or 1
  const int koff = grp * 8;
  int m = row0 + mrel;
  if (m >= Nn) m = Nn - 1;               // clamp loads; stores guarded below

  const v16h* Wf = (const v16h*)Wswz;    // fragment-major, 32B per lane-fragment

  v8f acc[NT];
  #pragma unroll
  for (int t = 0; t < NT; ++t)
    #pragma unroll
    for (int r = 0; r < 8; ++r) acc[t][r] = 0.0f;

  #pragma unroll
  for (int ksi = 0; ksi < K / 32; ++ksi) {
    const float4* xp = (const float4*)(X + (size_t)m * K + ksi * 32);
    float4 f0 = xp[(koff >> 2) + 0];
    float4 f1 = xp[(koff >> 2) + 1];
    float4 f2 = xp[((16 + koff) >> 2) + 0];
    float4 f3 = xp[((16 + koff) >> 2) + 1];
    v16h a;
    a[0]  = (_Float16)f0.x; a[1]  = (_Float16)f0.y; a[2]  = (_Float16)f0.z; a[3]  = (_Float16)f0.w;
    a[4]  = (_Float16)f1.x; a[5]  = (_Float16)f1.y; a[6]  = (_Float16)f1.z; a[7]  = (_Float16)f1.w;
    a[8]  = (_Float16)f2.x; a[9]  = (_Float16)f2.y; a[10] = (_Float16)f2.z; a[11] = (_Float16)f2.w;
    a[12] = (_Float16)f3.x; a[13] = (_Float16)f3.y; a[14] = (_Float16)f3.z; a[15] = (_Float16)f3.w;
    #pragma unroll
    for (int nt = 0; nt < NT; ++nt) {
      v16h b = Wf[((size_t)ksi * NT + nt) * 32 + lane];
      acc[nt] = __builtin_amdgcn_wmma_f32_16x16x32_f16(
          false, a, false, b, (short)0, acc[nt], false, false);
    }
  }

  const int mbase = row0 + grp * 8;
  #pragma unroll
  for (int nt = 0; nt < NT; ++nt)
    #pragma unroll
    for (int r = 0; r < 8; ++r) {
      int mr = mbase + r;
      if (mr < Nn) Y[(size_t)mr * NOUT + nt * 16 + mrel] = acc[nt][r];
    }
}

// a_src[n,h] = <h[n,h,:], att_src[h,:]>, likewise a_dst   (D = 32)
__global__ void attn_scalar_kernel(const float* __restrict__ H, int Nn, int heads,
                                   const float* __restrict__ att_s,
                                   const float* __restrict__ att_d,
                                   float* __restrict__ asrc, float* __restrict__ adst) {
  int i = blockIdx.x * blockDim.x + threadIdx.x;
  if (i >= Nn * heads) return;
  int n = i / heads, h = i - n * heads;
  const float* hr = H + (size_t)n * heads * 32 + h * 32;
  const float* as = att_s + h * 32;
  const float* ad = att_d + h * 32;
  float s = 0.f, d = 0.f;
  #pragma unroll
  for (int j = 0; j < 32; ++j) { float v = hr[j]; s += v * as[j]; d += v * ad[j]; }
  asrc[i] = s; adst[i] = d;
}

__device__ __forceinline__ void edge_nodes(int e, int E,
    const int* __restrict__ esrc, const int* __restrict__ edst, int& s, int& d) {
  if (e < E) { s = esrc[e]; d = edst[e]; } else { s = e - E; d = s; }  // self-loops
}

template<int H>
__global__ void edge_max_kernel(const int* __restrict__ esrc, const int* __restrict__ edst,
                                int E, int Etot,
                                const float* __restrict__ asrc, const float* __restrict__ adst,
                                float* __restrict__ emax) {
  int e = blockIdx.x * blockDim.x + threadIdx.x;
  if (e >= Etot) return;
  int s, d; edge_nodes(e, E, esrc, edst, s, d);
  #pragma unroll
  for (int h = 0; h < H; ++h) {
    float v = asrc[s * H + h] + adst[d * H + h];
    v = v > 0.f ? v : NEG_SLOPE * v;
    atomicMaxF(&emax[d * H + h], v);
  }
}

template<int H>
__global__ void edge_sum_kernel(const int* __restrict__ esrc, const int* __restrict__ edst,
                                int E, int Etot,
                                const float* __restrict__ asrc, const float* __restrict__ adst,
                                const float* __restrict__ emax, float* __restrict__ den) {
  int e = blockIdx.x * blockDim.x + threadIdx.x;
  if (e >= Etot) return;
  int s, d; edge_nodes(e, E, esrc, edst, s, d);
  #pragma unroll
  for (int h = 0; h < H; ++h) {
    float v = asrc[s * H + h] + adst[d * H + h];
    v = v > 0.f ? v : NEG_SLOPE * v;
    atomicAdd(&den[d * H + h], __expf(v - emax[d * H + h]));
  }
}

template<int H>
__global__ void edge_acc_kernel(const int* __restrict__ esrc, const int* __restrict__ edst,
                                int E, int Etot,
                                const float* __restrict__ asrc, const float* __restrict__ adst,
                                const float* __restrict__ emax, const float* __restrict__ den,
                                const float* __restrict__ Hf, float* __restrict__ acc) {
  int e = blockIdx.x * blockDim.x + threadIdx.x;
  if (e >= Etot) return;
  int s, d; edge_nodes(e, E, esrc, edst, s, d);
  #pragma unroll
  for (int h = 0; h < H; ++h) {
    float v = asrc[s * H + h] + adst[d * H + h];
    v = v > 0.f ? v : NEG_SLOPE * v;
    float alpha = __expf(v - emax[d * H + h]) / (den[d * H + h] + 1e-16f);
    const float4* hs = (const float4*)(Hf + (size_t)s * (H * 32) + h * 32);
    float* ap = acc + (size_t)d * (H * 32) + h * 32;
    #pragma unroll
    for (int q = 0; q < 8; ++q) {
      float4 hv = hs[q];
      atomicAdd(ap + q * 4 + 0, alpha * hv.x);
      atomicAdd(ap + q * 4 + 1, alpha * hv.y);
      atomicAdd(ap + q * 4 + 2, alpha * hv.z);
      atomicAdd(ap + q * 4 + 3, alpha * hv.w);
    }
  }
}

// h2in[n,d] = relu( (acc[n,d] + acc[n,32+d]) * 0.5 + b1[d] )
__global__ void mean_relu_kernel(const float* __restrict__ acc, const float* __restrict__ b1,
                                 float* __restrict__ out, int Nn) {
  int i = blockIdx.x * blockDim.x + threadIdx.x;
  if (i >= Nn * 32) return;
  int n = i >> 5, dd = i & 31;
  float v = 0.5f * (acc[(size_t)n * 64 + dd] + acc[(size_t)n * 64 + 32 + dd]) + b1[dd];
  out[i] = v > 0.f ? v : 0.f;
}

__global__ void finalize_kernel(float* __restrict__ embeds, const float* __restrict__ b2,
                                float* __restrict__ imp, int Nn) {
  int n = blockIdx.x * blockDim.x + threadIdx.x;
  if (n >= Nn) return;
  float ss = 0.f;
  #pragma unroll
  for (int dd = 0; dd < 32; ++dd) {
    float v = embeds[(size_t)n * 32 + dd] + b2[dd];
    embeds[(size_t)n * 32 + dd] = v;
    ss += v * v;
  }
  imp[n] = sqrtf(ss);
}

__global__ void topk_kernel(const float* __restrict__ imp, int Nn, float* __restrict__ out) {
  __shared__ float sv[256 * 5];
  __shared__ int   si[256 * 5];
  int t = threadIdx.x;
  float bv[5]; int bi[5];
  #pragma unroll
  for (int k = 0; k < 5; ++k) { bv[k] = -1.f; bi[k] = 0x7fffffff; }
  for (int i = t; i < Nn; i += 256) {
    float v = imp[i];
    if (v > bv[4] || (v == bv[4] && i < bi[4])) {
      bv[4] = v; bi[4] = i;
      #pragma unroll
      for (int k = 4; k > 0; --k) {
        if (bv[k] > bv[k - 1] || (bv[k] == bv[k - 1] && bi[k] < bi[k - 1])) {
          float tv = bv[k]; bv[k] = bv[k - 1]; bv[k - 1] = tv;
          int   ti = bi[k]; bi[k] = bi[k - 1]; bi[k - 1] = ti;
        } else break;
      }
    }
  }
  #pragma unroll
  for (int k = 0; k < 5; ++k) { sv[t * 5 + k] = bv[k]; si[t * 5 + k] = bi[k]; }
  __syncthreads();
  if (t == 0) {
    for (int k = 0; k < 5; ++k) {
      float best = -1.f; int besti = 0x7fffffff; int slot = 0;
      for (int j = 0; j < 256 * 5; ++j)
        if (sv[j] > best || (sv[j] == best && si[j] < besti)) { best = sv[j]; besti = si[j]; slot = j; }
      out[k] = (float)besti;
      sv[slot] = -2.f;
    }
  }
}

// ---------------------------------------------------------------------------
extern "C" void kernel_launch(void* const* d_in, const int* in_sizes, int n_in,
                              void* d_out, int out_size, void* d_ws, size_t ws_size,
                              hipStream_t stream) {
  (void)n_in; (void)out_size; (void)ws_size;
  const float* x   = (const float*)d_in[0];
  const int*   ei  = (const int*)d_in[1];
  const float* W1  = (const float*)d_in[2];
  const float* as1 = (const float*)d_in[3];
  const float* ad1 = (const float*)d_in[4];
  const float* b1  = (const float*)d_in[5];
  const float* W2  = (const float*)d_in[6];
  const float* as2 = (const float*)d_in[7];
  const float* ad2 = (const float*)d_in[8];
  const float* b2  = (const float*)d_in[9];

  const int Nn   = in_sizes[0] / 64;
  const int E    = in_sizes[1] / 2;
  const int Etot = E + Nn;
  const int* esrc = ei;
  const int* edst = ei + E;

  float* ws    = (float*)d_ws;
  float* h1    = ws;                       // N*64 fp32 (later reused as h2in, N*32)
  float* acc1  = h1 + (size_t)Nn * 64;     // N*64 fp32 (later reused as h2,   N*32)
  float* asrc1 = acc1 + (size_t)Nn * 64;   // N*2
  float* adst1 = asrc1 + (size_t)Nn * 2;   // N*2
  float* emax1 = adst1 + (size_t)Nn * 2;   // N*2
  float* den1  = emax1 + (size_t)Nn * 2;   // N*2
  float* asrc2 = den1 + (size_t)Nn * 2;    // N
  float* adst2 = asrc2 + Nn;               // N
  float* emax2 = adst2 + Nn;               // N
  float* den2  = emax2 + Nn;               // N
  _Float16* W1h = (_Float16*)(den2 + Nn);  // 64*64 halves, fragment-swizzled
  _Float16* W2h = W1h + 64 * 64;           // 32*32 halves, fragment-swizzled

  float* embeds = (float*)d_out;             // N*32
  float* imp    = embeds + (size_t)Nn * 32;  // N
  float* tidx   = imp + Nn;                  // 5

  const float NEGINF = -__builtin_inff();
  auto nb = [](int n, int b) { return (n + b - 1) / b; };

  // weight swizzle + init (re-done every call: harness poisons ws/d_out)
  swizzle_w_kernel<<<nb(64 * 64, 256), 256, 0, stream>>>(W1, W1h, 64, 64);
  swizzle_w_kernel<<<nb(32 * 32, 256), 256, 0, stream>>>(W2, W2h, 32, 32);
  fill_kernel<<<nb(Nn * 64, 256), 256, 0, stream>>>(acc1, Nn * 64, 0.f);
  fill_kernel<<<nb(Nn * 2, 256), 256, 0, stream>>>(emax1, Nn * 2, NEGINF);
  fill_kernel<<<nb(Nn * 2, 256), 256, 0, stream>>>(den1, Nn * 2, 0.f);
  fill_kernel<<<nb(Nn, 256), 256, 0, stream>>>(emax2, Nn, NEGINF);
  fill_kernel<<<nb(Nn, 256), 256, 0, stream>>>(den2, Nn, 0.f);
  fill_kernel<<<nb(Nn * 32, 256), 256, 0, stream>>>(embeds, Nn * 32, 0.f);

  // ---- layer 1 (heads=2, D=32) ----
  gemm_wmma_kernel<64, 64><<<nb(Nn, 16), 32, 0, stream>>>(x, W1h, h1, Nn);
  attn_scalar_kernel<<<nb(Nn * 2, 256), 256, 0, stream>>>(h1, Nn, 2, as1, ad1, asrc1, adst1);
  edge_max_kernel<2><<<nb(Etot, 256), 256, 0, stream>>>(esrc, edst, E, Etot, asrc1, adst1, emax1);
  edge_sum_kernel<2><<<nb(Etot, 256), 256, 0, stream>>>(esrc, edst, E, Etot, asrc1, adst1, emax1, den1);
  edge_acc_kernel<2><<<nb(Etot, 256), 256, 0, stream>>>(esrc, edst, E, Etot, asrc1, adst1,
                                                        emax1, den1, h1, acc1);
  mean_relu_kernel<<<nb(Nn * 32, 256), 256, 0, stream>>>(acc1, b1, h1, Nn);  // h2in -> h1 region

  // ---- layer 2 (heads=1, D=32) ----
  gemm_wmma_kernel<32, 32><<<nb(Nn, 16), 32, 0, stream>>>(h1, W2h, acc1, Nn); // h2 -> acc1 region
  attn_scalar_kernel<<<nb(Nn, 256), 256, 0, stream>>>(acc1, Nn, 1, as2, ad2, asrc2, adst2);
  edge_max_kernel<1><<<nb(Etot, 256), 256, 0, stream>>>(esrc, edst, E, Etot, asrc2, adst2, emax2);
  edge_sum_kernel<1><<<nb(Etot, 256), 256, 0, stream>>>(esrc, edst, E, Etot, asrc2, adst2, emax2, den2);
  edge_acc_kernel<1><<<nb(Etot, 256), 256, 0, stream>>>(esrc, edst, E, Etot, asrc2, adst2,
                                                        emax2, den2, acc1, embeds);

  finalize_kernel<<<nb(Nn, 256), 256, 0, stream>>>(embeds, b2, imp, Nn);
  topk_kernel<<<1, 256, 0, stream>>>(imp, Nn, tidx);
}
